// AttnProcessor_19430432047441
// MI455X (gfx1250) — compile-verified, matrix operand order checked by reading, not run on hardware
//
#include <hip/hip_runtime.h>
#include <hip/hip_bf16.h>

// ---------------------------------------------------------------------------
// Problem constants
// ---------------------------------------------------------------------------
#define B_    8
#define S_    1024
#define C_    640
#define H_    8
#define DH_   80
#define KV_   2048
#define M_    (B_ * S_)        // 8192 rows for all projections

typedef __attribute__((ext_vector_type(16))) __bf16 v16bf;
typedef __attribute__((ext_vector_type(8)))  float  v8f;
typedef __attribute__((ext_vector_type(4)))  unsigned int v4u;
typedef __attribute__((ext_vector_type(8)))  int v8i_;
typedef __attribute__((ext_vector_type(4)))  int v4i_;

union FragU {
    v16bf v;
    uint4 q[2];
    unsigned short u[16];
};

__device__ __forceinline__ unsigned short f2bf(float x) {
    unsigned int u = __float_as_uint(x);
    u += 0x7FFFu + ((u >> 16) & 1u);           // round-to-nearest-even
    return (unsigned short)(u >> 16);
}

__device__ __forceinline__ unsigned lds_lo32(const void* p) {
    return (unsigned)(unsigned long long)(size_t)p;     // LDS address is the pointer's low 32 bits
}

// ---------------------------------------------------------------------------
// CDNA5 async copy: global -> LDS (16B per lane), tracked by ASYNCcnt.
// ---------------------------------------------------------------------------
__device__ __forceinline__ void async_ld_b128(unsigned lds, const void* gptr) {
    asm volatile("global_load_async_to_lds_b128 %0, %1, off"
                 :: "v"(lds), "v"((unsigned long long)(size_t)gptr)
                 : "memory");
}
#define S_WAIT_ASYNC(imm) asm volatile("s_wait_asynccnt " imm ::: "memory")

// ---------------------------------------------------------------------------
// CDNA5 Tensor Data Mover: one 2-D tile DMA (global -> LDS), TENSORcnt.
// Tile: tile_d1 rows x tile_d0 elements (2-byte), row pitch `stride` elems.
// LDS receives the tile linearly (row-major, pitch = tile_d0).
// ---------------------------------------------------------------------------
__device__ __forceinline__ void tdm_load_2d(unsigned lds, const void* g,
                                            unsigned tensor_d0, unsigned tile_d0,
                                            unsigned tile_d1, unsigned stride) {
    unsigned long long ga = (unsigned long long)(size_t)g;
    v4u g0;
    g0[0] = 1u;                                              // count=1, user D#
    g0[1] = lds;                                             // lds_addr
    g0[2] = (unsigned)ga;                                    // global_addr[31:0]
    g0[3] = ((unsigned)(ga >> 32) & 0x01FFFFFFu) | 0x80000000u; // [56:32]|type=2
    v8i_ g1;
    g1[0] = 0x00010000;                                      // data_size = 2B
    g1[1] = (int)((tensor_d0 & 0xFFFFu) << 16);              // tensor_dim0 lo
    g1[2] = (int)((0x7FFFu << 16) | (tensor_d0 >> 16));      // dim1 lo | dim0 hi
    g1[3] = (int)(tile_d0 << 16);                            // tile_dim0 | dim1 hi=0
    g1[4] = (int)tile_d1;                                    // tile_dim1 (tile_dim2=0)
    g1[5] = (int)stride;                                     // tensor_dim0_stride lo
    g1[6] = 0;
    g1[7] = 0;
    v4i_ z4 = {0, 0, 0, 0};
    v8i_ z8 = {0, 0, 0, 0, 0, 0, 0, 0};
    __builtin_amdgcn_tensor_load_to_lds(g0, g1, z4, z4, z8, 0);
}

// Build a 16x32 bf16 A-style fragment from row-major global memory.
__device__ __forceinline__ v16bf load_frag(const unsigned short* __restrict__ base,
                                           int ld, long row, int kbase, bool partial) {
    FragU f;
    const uint4* p = (const uint4*)(base + row * (long)ld + kbase);
    f.q[0] = p[0];
    f.q[1] = partial ? make_uint4(0u, 0u, 0u, 0u) : p[2];
    return f.v;
}

// Same, from an LDS-resident row-major tile.
__device__ __forceinline__ v16bf frag_lds(const unsigned short* base,
                                          int row, int ld, int kbase, bool partial) {
    FragU f;
    const uint4* p = (const uint4*)(base + row * ld + kbase);
    f.q[0] = p[0];
    f.q[1] = partial ? make_uint4(0u, 0u, 0u, 0u) : p[2];
    return f.v;
}

__device__ __forceinline__ v8f vzero8() {
    v8f z = {0.f, 0.f, 0.f, 0.f, 0.f, 0.f, 0.f, 0.f};
    return z;
}

// ---------------------------------------------------------------------------
// Prep kernels
// ---------------------------------------------------------------------------
__global__ void k_f32_to_bf16(const float* __restrict__ in,
                              unsigned short* __restrict__ out, long n) {
    long i = (long)blockIdx.x * blockDim.x + threadIdx.x;
    if (i < n) out[i] = f2bf(in[i]);
}

__global__ void k_transpose_w(const float* __restrict__ W,
                              unsigned short* __restrict__ Wt) {
    int i = blockIdx.x * blockDim.x + threadIdx.x;
    if (i >= C_ * C_) return;
    int n = i / C_, k = i % C_;
    Wt[i] = f2bf(W[k * C_ + n]);
}

__global__ void k_ref_kv(const float* __restrict__ kref,
                         const float* __restrict__ vref,
                         const float* __restrict__ smask,
                         unsigned short* __restrict__ K_all,
                         unsigned short* __restrict__ V_allT) {
    long i = (long)blockIdx.x * blockDim.x + threadIdx.x;
    if (i >= (long)B_ * S_ * C_) return;
    int c = (int)(i % C_);
    long r = i / C_;
    int s = (int)(r % S_);
    int b = (int)(r / S_);
    int f = b & 3;
    float bg = 1.f - smask[f * 65536 + (s >> 5) * 2048 + (s & 31) * 8];
    long src = ((long)f * S_ + s) * C_ + c;
    K_all [((long)b * KV_ + S_ + s) * C_ + c]   = f2bf(kref[src] * bg);
    V_allT[((long)b * C_ + c) * KV_ + S_ + s]   = f2bf(vref[src] * bg);
}

// ---------------------------------------------------------------------------
// bf16 WMMA GEMM: out = X[8192x640] @ W[640x640]
// Weight tile double-buffered in LDS via async-to-LDS; main loop unrolled x2
// so buffer identity is compile-time and accumulators chain in place.
// A fragment software-pipelined one k-step ahead (alternating registers).
// ---------------------------------------------------------------------------
__global__ __launch_bounds__(128) void k_gemm_bf16(
        const unsigned short* __restrict__ X,
        const unsigned short* __restrict__ Wt,
        int mode,
        unsigned short* __restrict__ outb,
        float* __restrict__ outf,
        const float* __restrict__ mask,
        const float* __restrict__ bias) {
    __shared__ unsigned short ldsW[2][64 * 32];   // 2 x 4KB double buffer

    const int tid  = threadIdx.x;
    const int wave = tid >> 5;
    const int lane = tid & 31;
    const int half = lane >> 4;
    const int lr   = lane & 15;
    const int m0 = (blockIdx.x * 4 + wave) * 16;
    const int n0 = blockIdx.y * 64;

    v8f acc[4];
#pragma unroll
    for (int t = 0; t < 4; ++t) acc[t] = vzero8();

    auto stage_w = [&](int k0, unsigned short* buf) {
#pragma unroll
        for (int j = 0; j < 2; ++j) {
            int t   = tid * 2 + j;
            int row = t >> 2, ch = t & 3;
            async_ld_b128(lds_lo32(buf + row * 32 + ch * 8),
                          Wt + (long)(n0 + row) * C_ + k0 + ch * 8);
        }
    };
    auto compute = [&](const unsigned short* buf, v16bf a) {
        v16bf bfr[4];
#pragma unroll
        for (int t = 0; t < 4; ++t)
            bfr[t] = frag_lds(buf, t * 16 + lr, 32, (half << 3), false);
#pragma unroll
        for (int t = 0; t < 4; ++t)
            acc[t] = __builtin_amdgcn_wmma_f32_16x16x32_bf16(
                false, a, false, bfr[t], (short)0, acc[t], false, false);
    };

    stage_w(0, ldsW[0]);
    v16bf a0 = load_frag(X, C_, m0 + lr, (half << 3), false);
    v16bf a1;

#pragma unroll 1
    for (int ks = 0; ks < 16; ks += 2) {
        stage_w((ks + 1) * 32, ldsW[1]);
        a1 = load_frag(X, C_, m0 + lr, (ks + 1) * 32 + (half << 3), false);
        S_WAIT_ASYNC("0x2");
        __syncthreads();
        compute(ldsW[0], a0);
        __syncthreads();

        stage_w((ks + 2) * 32, ldsW[0]);
        a0 = load_frag(X, C_, m0 + lr, (ks + 2) * 32 + (half << 3), false);
        S_WAIT_ASYNC("0x2");
        __syncthreads();
        compute(ldsW[1], a1);
        __syncthreads();
    }
    // ks = 16,17 (prefetch 17, 18)
    stage_w(17 * 32, ldsW[1]);
    a1 = load_frag(X, C_, m0 + lr, 17 * 32 + (half << 3), false);
    S_WAIT_ASYNC("0x2"); __syncthreads();
    compute(ldsW[0], a0); __syncthreads();
    stage_w(18 * 32, ldsW[0]);
    a0 = load_frag(X, C_, m0 + lr, 18 * 32 + (half << 3), false);
    S_WAIT_ASYNC("0x2"); __syncthreads();
    compute(ldsW[1], a1); __syncthreads();
    // ks = 18 (prefetch 19), ks = 19 (drain)
    stage_w(19 * 32, ldsW[1]);
    a1 = load_frag(X, C_, m0 + lr, 19 * 32 + (half << 3), false);
    S_WAIT_ASYNC("0x2"); __syncthreads();
    compute(ldsW[0], a0); __syncthreads();
    S_WAIT_ASYNC("0x0"); __syncthreads();
    compute(ldsW[1], a1);

    if (mode == 3) {
#pragma unroll
        for (int t = 0; t < 4; ++t) {
            int n = n0 + t * 16 + lr;
            float bo = bias[n];
#pragma unroll
            for (int i = 0; i < 8; ++i) {
                int m = m0 + i + (half << 3);
                outf[(long)m * C_ + n] = acc[t][i] + bo;
            }
        }
    } else if (mode == 0) {
#pragma unroll
        for (int t = 0; t < 4; ++t) {
            int n = n0 + t * 16 + lr;
#pragma unroll
            for (int i = 0; i < 8; ++i) {
                int m = m0 + i + (half << 3);
                outb[(long)m * C_ + n] = f2bf(acc[t][i]);
            }
        }
    } else {
#pragma unroll
        for (int i = 0; i < 8; ++i) {
            int m = m0 + i + (half << 3);
            int b = m >> 10, s = m & 1023, f = b & 3;
            float fg = mask[f * 65536 + (s >> 5) * 2048 + (s & 31) * 8];
#pragma unroll
            for (int t = 0; t < 4; ++t) {
                int n = n0 + t * 16 + lr;
                unsigned short val = f2bf(acc[t][i] * fg);
                if (mode == 1)
                    outb[((long)b * KV_ + s) * C_ + n] = val;            // K_all
                else
                    outb[((long)b * C_ + n) * KV_ + s] = val;            // V_allT
            }
        }
    }
}

// ---------------------------------------------------------------------------
// Flash attention: 1 block = 4 waves = 64 q rows of one (b,h).
// K tile (64x80) / V tile (80x64) staged by the Tensor Data Mover (one
// tensor_load_to_lds per tile, wave 0 only), double-buffered, prefetch
// distance 1, drained with s_wait_tensorcnt. Compute loop unrolled x2 so
// buffer identity is static.
// ---------------------------------------------------------------------------
__global__ __launch_bounds__(128) void k_attention(
        const unsigned short* __restrict__ Qb,      // [8192,640] bf16
        const unsigned short* __restrict__ K_all,   // [8,2048,640] bf16
        const unsigned short* __restrict__ V_allT,  // [8,640,2048] bf16
        unsigned short* __restrict__ outb) {        // [8192,640] bf16
    __shared__ unsigned short ldsP[4 * 16 * 64];    //  8 KB
    __shared__ unsigned short ldsK[2][64 * 80];     // 20 KB (kv-major, pitch 80)
    __shared__ unsigned short ldsV[2][80 * 64];     // 20 KB (ch-major, pitch 64)

    const int tid  = threadIdx.x;
    const int wave = tid >> 5;
    const int lane = tid & 31;
    const int half = lane >> 4;
    const int lr   = lane & 15;
    const int b  = blockIdx.y >> 3;
    const int h  = blockIdx.y & 7;
    const int q0 = blockIdx.x * 64 + wave * 16;
    const int hb = h * DH_;
    unsigned short* Pw = ldsP + wave * 16 * 64;

    // One TDM descriptor per K tile and per V tile, issued by wave 0 only.
    auto stage_kv = [&](int kv0, unsigned short* kbuf, unsigned short* vbuf) {
        if (wave == 0) {
            tdm_load_2d(lds_lo32(kbuf),
                        K_all + ((long)b * KV_ + kv0) * C_ + hb,
                        C_,  /*tile_d0=*/DH_, /*tile_d1=*/64, /*stride=*/C_);
            tdm_load_2d(lds_lo32(vbuf),
                        V_allT + ((long)b * C_ + hb) * KV_ + kv0,
                        KV_, /*tile_d0=*/64,  /*tile_d1=*/DH_, /*stride=*/KV_);
        }
    };

    // Preload Q fragments (3 k-steps of 32 covering dh=80, last padded)
    v16bf qf[3];
#pragma unroll
    for (int kk = 0; kk < 3; ++kk)
        qf[kk] = load_frag(Qb, C_, (long)b * S_ + q0 + lr,
                           hb + kk * 32 + (half << 3), kk == 2);

    float m[8], l[8];
#pragma unroll
    for (int i = 0; i < 8; ++i) { m[i] = -3.0e38f; l[i] = 0.f; }
    v8f o[5];
#pragma unroll
    for (int t = 0; t < 5; ++t) o[t] = vzero8();

    const float scale = 0.1118033988749895f;   // 1/sqrt(80)

    // process one staged kv tile (scores -> online softmax -> PV)
    auto attn_tile = [&](const unsigned short* kbuf, const unsigned short* vbuf) {
        v8f sc[4];
#pragma unroll
        for (int t = 0; t < 4; ++t) sc[t] = vzero8();
#pragma unroll
        for (int t = 0; t < 4; ++t) {
            v16bf kfr[3];
#pragma unroll
            for (int kk = 0; kk < 3; ++kk)
                kfr[kk] = frag_lds(kbuf, t * 16 + lr, 80,
                                   kk * 32 + (half << 3), kk == 2);
#pragma unroll
            for (int kk = 0; kk < 3; ++kk)
                sc[t] = __builtin_amdgcn_wmma_f32_16x16x32_bf16(
                    false, qf[kk], false, kfr[kk], (short)0, sc[t], false, false);
        }

#pragma unroll
        for (int i = 0; i < 8; ++i) {
            float s0 = sc[0][i] * scale, s1 = sc[1][i] * scale;
            float s2 = sc[2][i] * scale, s3 = sc[3][i] * scale;
            float mx = fmaxf(fmaxf(s0, s1), fmaxf(s2, s3));
#pragma unroll
            for (int d = 1; d < 16; d <<= 1)
                mx = fmaxf(mx, __shfl_xor(mx, d, 32));
            float mnew = fmaxf(m[i], mx);
            float corr = __expf(m[i] - mnew);
            m[i] = mnew;
            float p0 = __expf(s0 - mnew), p1 = __expf(s1 - mnew);
            float p2 = __expf(s2 - mnew), p3 = __expf(s3 - mnew);
            float rs = p0 + p1 + p2 + p3;
#pragma unroll
            for (int d = 1; d < 16; d <<= 1)
                rs += __shfl_xor(rs, d, 32);
            l[i] = l[i] * corr + rs;
            int r = i + (half << 3);
            Pw[r * 64 +  0 + lr] = f2bf(p0);
            Pw[r * 64 + 16 + lr] = f2bf(p1);
            Pw[r * 64 + 32 + lr] = f2bf(p2);
            Pw[r * 64 + 48 + lr] = f2bf(p3);
#pragma unroll
            for (int t = 0; t < 5; ++t) o[t][i] *= corr;
        }

        __syncthreads();                    // P visible across lanes

#pragma unroll
        for (int kk = 0; kk < 2; ++kk) {
            v16bf pA = frag_lds(Pw, lr, 64, kk * 32 + (half << 3), false);
            v16bf vb[5];
#pragma unroll
            for (int t = 0; t < 5; ++t)
                vb[t] = frag_lds(vbuf, t * 16 + lr, 64,
                                 kk * 32 + (half << 3), false);
#pragma unroll
            for (int t = 0; t < 5; ++t)
                o[t] = __builtin_amdgcn_wmma_f32_16x16x32_bf16(
                    false, pA, false, vb[t], (short)0, o[t], false, false);
        }

        __syncthreads();                    // tile reads done; buffers free
    };

    stage_kv(0, ldsK[0], ldsV[0]);

#pragma unroll 1
    for (int kt = 0; kt < 30; kt += 2) {
        stage_kv((kt + 1) * 64, ldsK[1], ldsV[1]);
        __builtin_amdgcn_s_wait_tensorcnt(2);
        __syncthreads();
        attn_tile(ldsK[0], ldsV[0]);

        stage_kv((kt + 2) * 64, ldsK[0], ldsV[0]);
        __builtin_amdgcn_s_wait_tensorcnt(2);
        __syncthreads();
        attn_tile(ldsK[1], ldsV[1]);
    }
    // kt = 30 (prefetch 31), kt = 31 (drain)
    stage_kv(31 * 64, ldsK[1], ldsV[1]);
    __builtin_amdgcn_s_wait_tensorcnt(2);
    __syncthreads();
    attn_tile(ldsK[0], ldsV[0]);
    __builtin_amdgcn_s_wait_tensorcnt(0);
    __syncthreads();
    attn_tile(ldsK[1], ldsV[1]);

    // --- normalize and store ---
#pragma unroll
    for (int i = 0; i < 8; ++i) {
        float inv = 1.f / l[i];
        long grow = (long)b * S_ + q0 + i + (half << 3);
#pragma unroll
        for (int t = 0; t < 5; ++t)
            outb[grow * C_ + hb + t * 16 + lr] = f2bf(o[t][i] * inv);
    }
}

// ---------------------------------------------------------------------------
// Launch
// ---------------------------------------------------------------------------
extern "C" void kernel_launch(void* const* d_in, const int* in_sizes, int n_in,
                              void* d_out, int out_size, void* d_ws, size_t ws_size,
                              hipStream_t stream) {
    const float* hs    = (const float*)d_in[0];
    const float* Wq    = (const float*)d_in[1];
    const float* Wk    = (const float*)d_in[2];
    const float* Wv    = (const float*)d_in[3];
    const float* Wo    = (const float*)d_in[4];
    const float* bo    = (const float*)d_in[5];
    const float* kref  = (const float*)d_in[6];
    const float* vref  = (const float*)d_in[7];
    const float* smask = (const float*)d_in[8];
    const float* tmask = (const float*)d_in[9];
    float* out = (float*)d_out;

    char* p = (char*)d_ws;
    unsigned short* hsb   = (unsigned short*)p; p += (long)M_ * C_ * 2;
    unsigned short* WqT   = (unsigned short*)p; p += (long)C_ * C_ * 2;
    unsigned short* WkT   = (unsigned short*)p; p += (long)C_ * C_ * 2;
    unsigned short* WvT   = (unsigned short*)p; p += (long)C_ * C_ * 2;
    unsigned short* WoT   = (unsigned short*)p; p += (long)C_ * C_ * 2;
    unsigned short* Qb    = (unsigned short*)p; p += (long)M_ * C_ * 2;
    unsigned short* K_all = (unsigned short*)p; p += (long)B_ * KV_ * C_ * 2;
    unsigned short* V_allT= (unsigned short*)p; p += (long)B_ * C_ * KV_ * 2;
    unsigned short* attnb = (unsigned short*)p; p += (long)M_ * C_ * 2;

    const long nHS = (long)M_ * C_;
    k_f32_to_bf16<<<(int)((nHS + 255) / 256), 256, 0, stream>>>(hs, hsb, nHS);
    k_transpose_w<<<(C_ * C_ + 255) / 256, 256, 0, stream>>>(Wq, WqT);
    k_transpose_w<<<(C_ * C_ + 255) / 256, 256, 0, stream>>>(Wk, WkT);
    k_transpose_w<<<(C_ * C_ + 255) / 256, 256, 0, stream>>>(Wv, WvT);
    k_transpose_w<<<(C_ * C_ + 255) / 256, 256, 0, stream>>>(Wo, WoT);
    k_ref_kv<<<(int)((nHS + 255) / 256), 256, 0, stream>>>(kref, vref, smask,
                                                           K_all, V_allT);

    dim3 gg(M_ / 64, C_ / 64);   // 128 x 10
    k_gemm_bf16<<<gg, 128, 0, stream>>>(hsb, WqT, 0, Qb,    nullptr, nullptr, nullptr);
    k_gemm_bf16<<<gg, 128, 0, stream>>>(hsb, WkT, 1, K_all, nullptr, tmask,   nullptr);
    k_gemm_bf16<<<gg, 128, 0, stream>>>(hsb, WvT, 2, V_allT,nullptr, tmask,   nullptr);

    k_attention<<<dim3(S_ / 64, B_ * H_), 128, 0, stream>>>(Qb, K_all, V_allT, attnb);

    k_gemm_bf16<<<gg, 128, 0, stream>>>(attnb, WoT, 3, nullptr, out, nullptr, bo);
}
